// KNNAttention_43928925503562
// MI455X (gfx1250) — matrix operand order repaired and checked by reading, not compiled, not run on hardware
//
#include <hip/hip_runtime.h>
#include <hip/hip_fp16.h>
#include <cfloat>
#include <cmath>

typedef __attribute__((ext_vector_type(16))) _Float16 v16h;
typedef __attribute__((ext_vector_type(8)))  _Float16 v8h;
typedef __attribute__((ext_vector_type(8)))  float    v8f;

#define HH    8
#define DDIM  64
#define KMEM  16
#define BB    2
#define NN    2048
#define DIM   1024
#define INNER 512

static __device__ __forceinline__ v16h cat8(v8h lo, v8h hi) {
    return __builtin_shufflevector(lo, hi, 0,1,2,3,4,5,6,7,8,9,10,11,12,13,14,15);
}
static __device__ __forceinline__ v8f wmma16(v16h a, v16h b, v8f c) {
    // D = A(16x32 f16) * B(32x16 f16) + C(16x16 f32)
    return __builtin_amdgcn_wmma_f32_16x16x32_f16(false, a, false, b, (short)0, c, false, false);
}

// ---------------------------------------------------------------------------
// Shared GEMM tile machinery: block 256 = 8 waves, tile 64(M) x 128(N),
// K staged in steps of 32 through padded LDS (row stride 40 halves = 80 B,
// keeping per-lane b128 fragment reads 16B-aligned and bank-conflict-free).
// ---------------------------------------------------------------------------
#define GEMM_STAGE_W()                                                           \
    _Pragma("unroll")                                                            \
    for (int j = 0; j < 4; ++j) {                                                \
        int f4 = t + 256 * j;                                                    \
        int kr = f4 >> 5, n4 = (f4 & 31) * 4;                                    \
        float4 wv = *reinterpret_cast<const float4*>(                            \
            &W[(size_t)(k0 + kr) * N + n1 + n4]);                                \
        sW[(n4 + 0) * 40 + kr] = (_Float16)wv.x;                                 \
        sW[(n4 + 1) * 40 + kr] = (_Float16)wv.y;                                 \
        sW[(n4 + 2) * 40 + kr] = (_Float16)wv.z;                                 \
        sW[(n4 + 3) * 40 + kr] = (_Float16)wv.w;                                 \
    }

#define GEMM_FRAGMENTS_AND_MMA()                                                 \
    const int kg8 = hi * 8, kg16 = hi * 16;                                      \
    v8h alo = *reinterpret_cast<const v8h*>(&sX[(msub + lm) * 40 + kg8]);        \
    v8h ahi = *reinterpret_cast<const v8h*>(&sX[(msub + lm) * 40 + 16 + kg8]);   \
    v16h a  = cat8(alo, ahi);                                                    \
    v16h bfr[4];                                                                 \
    _Pragma("unroll")                                                            \
    for (int f = 0; f < 4; ++f) {                                                \
        int n = nsub + f * 16 + lm;                                              \
        v8h blo = *reinterpret_cast<const v8h*>(&sW[n * 40 + kg8]);              \
        v8h bhi = *reinterpret_cast<const v8h*>(&sW[n * 40 + 16 + kg8]);         \
        bfr[f]  = cat8(blo, bhi);                                                \
    }                                                                            \
    _Pragma("unroll")                                                            \
    for (int f = 0; f < 4; ++f) acc[f] = wmma16(a, bfr[f], acc[f]);

// ---------------------------------------------------------------------------
// q projection fused with per-head l2-norm: Qh(f16) = l2norm_head(X @ Wq)
// Each wave's 16x64 fragment is one head's row slice -> in-register norm.
// ---------------------------------------------------------------------------
__global__ __launch_bounds__(256) void gemm_qnorm(
    const float* __restrict__ X, const float* __restrict__ W,
    _Float16* __restrict__ Qh, int M, int N, int K)
{
    __shared__ __align__(16) _Float16 sX[64 * 40];
    __shared__ __align__(16) _Float16 sW[128 * 40];
    const int t    = threadIdx.x;
    const int lane = t & 31, w = t >> 5;
    const int lm   = lane & 15, hi = lane >> 4;
    const int m0   = blockIdx.y * 64;
    const int n1   = blockIdx.x * 128;
    const int msub = (w & 3) * 16, nsub = (w >> 2) * 64;

    v8f acc[4] = {};
    for (int k0 = 0; k0 < K; k0 += 32) {
        #pragma unroll
        for (int j = 0; j < 2; ++j) {
            int f4  = t + 256 * j;
            int row = f4 >> 3, k4 = (f4 & 7) * 4;
            float4 xv = *reinterpret_cast<const float4*>(&X[(size_t)(m0 + row) * K + k0 + k4]);
            sX[row * 40 + k4 + 0] = (_Float16)xv.x;
            sX[row * 40 + k4 + 1] = (_Float16)xv.y;
            sX[row * 40 + k4 + 2] = (_Float16)xv.z;
            sX[row * 40 + k4 + 3] = (_Float16)xv.w;
        }
        GEMM_STAGE_W();
        __syncthreads();
        GEMM_FRAGMENTS_AND_MMA();
        __syncthreads();
    }
    // fused l2-norm over the head dimension (64 cols owned by this wave)
    #pragma unroll
    for (int r = 0; r < 8; ++r) {
        float ssq = 0.0f;
        #pragma unroll
        for (int f = 0; f < 4; ++f) ssq += acc[f][r] * acc[f][r];
        #pragma unroll
        for (int msk = 1; msk < 16; msk <<= 1) ssq += __shfl_xor(ssq, msk);
        float inv = 1.0f / fmaxf(sqrtf(ssq), 1e-12f);
        size_t rb = (size_t)(m0 + msub + r + 8 * hi) * N + n1 + nsub;
        #pragma unroll
        for (int f = 0; f < 4; ++f)
            Qh[rb + f * 16 + lm] = (_Float16)(acc[f][r] * inv);
    }
}

// ---------------------------------------------------------------------------
// kv projection fused with split: cols 0-63 -> l2norm -> Kh, cols 64-127 -> Vh
// ---------------------------------------------------------------------------
__global__ __launch_bounds__(256) void gemm_kvnorm(
    const float* __restrict__ X, const float* __restrict__ W,
    _Float16* __restrict__ Kh, _Float16* __restrict__ Vh, int M, int N, int K)
{
    __shared__ __align__(16) _Float16 sX[64 * 40];
    __shared__ __align__(16) _Float16 sW[128 * 40];
    const int t    = threadIdx.x;
    const int lane = t & 31, w = t >> 5;
    const int lm   = lane & 15, hi = lane >> 4;
    const int m0   = blockIdx.y * 64;
    const int n1   = 0;
    const int msub = (w & 3) * 16, nsub = (w >> 2) * 64;

    v8f acc[4] = {};
    for (int k0 = 0; k0 < K; k0 += 32) {
        #pragma unroll
        for (int j = 0; j < 2; ++j) {
            int f4  = t + 256 * j;
            int row = f4 >> 3, k4 = (f4 & 7) * 4;
            float4 xv = *reinterpret_cast<const float4*>(&X[(size_t)(m0 + row) * K + k0 + k4]);
            sX[row * 40 + k4 + 0] = (_Float16)xv.x;
            sX[row * 40 + k4 + 1] = (_Float16)xv.y;
            sX[row * 40 + k4 + 2] = (_Float16)xv.z;
            sX[row * 40 + k4 + 3] = (_Float16)xv.w;
        }
        GEMM_STAGE_W();
        __syncthreads();
        GEMM_FRAGMENTS_AND_MMA();
        __syncthreads();
    }
    if (nsub == 0) {        // k: normalize
        #pragma unroll
        for (int r = 0; r < 8; ++r) {
            float ssq = 0.0f;
            #pragma unroll
            for (int f = 0; f < 4; ++f) ssq += acc[f][r] * acc[f][r];
            #pragma unroll
            for (int msk = 1; msk < 16; msk <<= 1) ssq += __shfl_xor(ssq, msk);
            float inv = 1.0f / fmaxf(sqrtf(ssq), 1e-12f);
            size_t rb = (size_t)(m0 + msub + r + 8 * hi) * DDIM;
            #pragma unroll
            for (int f = 0; f < 4; ++f)
                Kh[rb + f * 16 + lm] = (_Float16)(acc[f][r] * inv);
        }
    } else {                // v: passthrough convert
        #pragma unroll
        for (int r = 0; r < 8; ++r) {
            size_t rb = (size_t)(m0 + msub + r + 8 * hi) * DDIM;
            #pragma unroll
            for (int f = 0; f < 4; ++f)
                Vh[rb + f * 16 + lm] = (_Float16)acc[f][r];
        }
    }
}

// ---------------------------------------------------------------------------
// Output GEMM: Out = xres + (Xh(f16) @ W) * tanh(gate)
// ---------------------------------------------------------------------------
__global__ __launch_bounds__(256) void gemm_out(
    const _Float16* __restrict__ Xh, const float* __restrict__ W,
    const float* __restrict__ xres, const float* __restrict__ gate,
    float* __restrict__ Out, int M, int N, int K)
{
    __shared__ __align__(16) _Float16 sX[64 * 40];
    __shared__ __align__(16) _Float16 sW[128 * 40];
    const int t    = threadIdx.x;
    const int lane = t & 31, w = t >> 5;
    const int lm   = lane & 15, hi = lane >> 4;
    const int m0   = blockIdx.y * 64;
    const int n1   = blockIdx.x * 128;
    const int msub = (w & 3) * 16, nsub = (w >> 2) * 64;

    v8f acc[4] = {};
    for (int k0 = 0; k0 < K; k0 += 32) {
        {   // X tile already f16: 2048 halves, 8 per thread
            int row = t >> 2, k8 = (t & 3) * 8;
            v8h xv = *reinterpret_cast<const v8h*>(&Xh[(size_t)(m0 + row) * K + k0 + k8]);
            *reinterpret_cast<v8h*>(&sX[row * 40 + k8]) = xv;
        }
        GEMM_STAGE_W();
        __syncthreads();
        GEMM_FRAGMENTS_AND_MMA();
        __syncthreads();
    }
    const float g = tanhf(gate[0]);
    #pragma unroll
    for (int f = 0; f < 4; ++f)
        #pragma unroll
        for (int r = 0; r < 8; ++r) {
            size_t idx = (size_t)(m0 + msub + r + 8 * hi) * N + n1 + nsub + f * 16 + lm;
            Out[idx] = xres[idx] + acc[f][r] * g;
        }
}

// ---------------------------------------------------------------------------
// V-tile staging helper (32 keys x 64 feats, transposed to [feat][key])
// ---------------------------------------------------------------------------
static __device__ __forceinline__ void stage_v(
    const _Float16* __restrict__ Vh, _Float16* sVbuf, int rowbase, int t)
{
    int key = t >> 3, f8 = (t & 7) * 8;
    v8h vv = *reinterpret_cast<const v8h*>(&Vh[(size_t)(rowbase + key) * DDIM + f8]);
    #pragma unroll
    for (int c = 0; c < 8; ++c) sVbuf[(f8 + c) * 40 + key] = vv[c];
}

// ---------------------------------------------------------------------------
// Causal local attention, flash-style. One wave per 16-query tile; 8 waves
// per block share a double-buffered LDS V tile (one barrier per iteration).
// Outputs unnormalized accumulator Acc(bh,n,64) plus per-row (m,l) stats.
// ---------------------------------------------------------------------------
__global__ __launch_bounds__(256) void flash_local(
    const _Float16* __restrict__ Qh, const _Float16* __restrict__ Kh,
    const _Float16* __restrict__ Vh, const float* __restrict__ scale_param,
    float* __restrict__ Acc, float* __restrict__ Mst, float* __restrict__ Lst)
{
    __shared__ __align__(16) _Float16 sV[2][64 * 40];   // double-buffered, [feat][key]
    __shared__ __align__(16) _Float16 sP[8][16 * 40];   // per-wave P transpose staging
    const int t = threadIdx.x, lane = t & 31, w = t >> 5;
    const int lm = lane & 15, hi = lane >> 4;
    const int bh = blockIdx.y, bi = bh / HH, h = bh % HH;
    const int q0 = blockIdx.x * 128 + w * 16;
    const float scale = expf(scale_param[h]);

    // Q A-fragments (feat groups 0-31, 32-63), loaded once
    const _Float16* qrow = Qh + (size_t)(bi * NN + q0 + lm) * INNER + h * DDIM;
    v16h aQ[2];
    #pragma unroll
    for (int g = 0; g < 2; ++g) {
        v8h lo = *reinterpret_cast<const v8h*>(qrow + g * 32 + hi * 8);
        v8h hg = *reinterpret_cast<const v8h*>(qrow + g * 32 + 16 + hi * 8);
        aQ[g] = cat8(lo, hg);
    }
    float m_st[8], l_st[8];
    #pragma unroll
    for (int r = 0; r < 8; ++r) { m_st[r] = -FLT_MAX; l_st[r] = 0.0f; }
    v8f o[4] = {};

    const int ntile = blockIdx.x * 4 + 4;       // key tiles up to block's max query
    stage_v(Vh, sV[0], bi * NN, t);
    __syncthreads();
    int buf = 0;
    for (int kt = 0; kt < ntile; ++kt) {
        const int j0 = kt * 32;
        if (kt + 1 < ntile) stage_v(Vh, sV[buf ^ 1], bi * NN + j0 + 32, t);

        // load all K fragments first, then issue the 4 S-WMMAs
        v16h bK[2][2];
        #pragma unroll
        for (int sub = 0; sub < 2; ++sub) {
            const _Float16* krow = Kh + (size_t)(bi * NN + j0 + sub * 16 + lm) * DDIM;
            #pragma unroll
            for (int g = 0; g < 2; ++g) {
                v8h lo = *reinterpret_cast<const v8h*>(krow + g * 32 + hi * 8);
                v8h hg = *reinterpret_cast<const v8h*>(krow + g * 32 + 16 + hi * 8);
                bK[sub][g] = cat8(lo, hg);
            }
        }
        v8f s0v = {}, s1v = {};
        s0v = wmma16(aQ[0], bK[0][0], s0v);
        s1v = wmma16(aQ[0], bK[1][0], s1v);
        s0v = wmma16(aQ[1], bK[0][1], s0v);
        s1v = wmma16(aQ[1], bK[1][1], s1v);

        // scale + causal mask + online softmax (row reductions over 16 lanes)
        #pragma unroll
        for (int r = 0; r < 8; ++r) {
            const int qi = q0 + r + 8 * hi;
            float s0 = s0v[r] * scale;
            float s1 = s1v[r] * scale;
            if (j0 + lm      > qi) s0 = -FLT_MAX;
            if (j0 + 16 + lm > qi) s1 = -FLT_MAX;
            float mx = fmaxf(s0, s1);
            #pragma unroll
            for (int msk = 1; msk < 16; msk <<= 1) mx = fmaxf(mx, __shfl_xor(mx, msk));
            float newm  = fmaxf(m_st[r], mx);
            float alpha = expf(m_st[r] - newm);
            float p0 = expf(s0 - newm);
            float p1 = expf(s1 - newm);
            float rs = p0 + p1;
            #pragma unroll
            for (int msk = 1; msk < 16; msk <<= 1) rs += __shfl_xor(rs, msk);
            l_st[r] = l_st[r] * alpha + rs;
            m_st[r] = newm;
            #pragma unroll
            for (int f = 0; f < 4; ++f) o[f][r] *= alpha;
            const int m = r + 8 * hi;
            sP[w][m * 40 + lm]      = (_Float16)p0;
            sP[w][m * 40 + 16 + lm] = (_Float16)p1;
        }
        asm volatile("s_wait_dscnt 0" ::: "memory");  // same-wave LDS RAW fence

        // P as A-fragment (16 queries x 32 keys); batch-load bV then 4 WMMAs
        v8h plo = *reinterpret_cast<const v8h*>(&sP[w][lm * 40 + hi * 8]);
        v8h phi = *reinterpret_cast<const v8h*>(&sP[w][lm * 40 + 16 + hi * 8]);
        v16h aP = cat8(plo, phi);
        const _Float16* sVb = sV[buf];
        v16h bV[4];
        #pragma unroll
        for (int f = 0; f < 4; ++f) {
            int nf = f * 16 + lm;
            v8h blo = *reinterpret_cast<const v8h*>(&sVb[nf * 40 + hi * 8]);
            v8h bhi = *reinterpret_cast<const v8h*>(&sVb[nf * 40 + 16 + hi * 8]);
            bV[f] = cat8(blo, bhi);
        }
        #pragma unroll
        for (int f = 0; f < 4; ++f) o[f] = wmma16(aP, bV[f], o[f]);

        __syncthreads();
        buf ^= 1;
    }

    #pragma unroll
    for (int f = 0; f < 4; ++f)
        #pragma unroll
        for (int r = 0; r < 8; ++r)
            Acc[((size_t)bh * NN + q0 + r + 8 * hi) * DDIM + f * 16 + lm] = o[f][r];
    if (lm == 0) {
        #pragma unroll
        for (int r = 0; r < 8; ++r) {
            Mst[(size_t)bh * NN + q0 + r + 8 * hi] = m_st[r];
            Lst[(size_t)bh * NN + q0 + r + 8 * hi] = l_st[r];
        }
    }
}

// ---------------------------------------------------------------------------
// Memory-KV attention: one wave per (b,h,i). Streams the 268 MB mem_kv once
// (the bandwidth-dominant pass); prefetch the value half during the k-dot.
// ---------------------------------------------------------------------------
__global__ __launch_bounds__(256) void mem_attn(
    const _Float16* __restrict__ Qh, const float* __restrict__ mem_kv,
    const unsigned char* __restrict__ mem_mask, const float* __restrict__ scale_param,
    float* __restrict__ Acc, float* __restrict__ Mst, float* __restrict__ Lst)
{
    const int t = threadIdx.x, lane = t & 31, w = t >> 5;
    const size_t gw = (size_t)blockIdx.x * 8 + w;   // 0 .. b*H*n-1
    const int i  = (int)(gw % NN);
    const int bh = (int)(gw / NN);
    const int h  = bh % HH, bi = bh / HH;
    const float scale = expf(scale_param[h]);
    const int key = lane >> 1, half = lane & 1;
    const size_t base = gw * (size_t)(KMEM * 2 * DDIM);

    // prefetch this row's value half while we compute the key dots
    __builtin_prefetch(mem_kv + base + (size_t)key * (2 * DDIM) + DDIM + half * 32, 0, 1);

    // sim[key] = scale * q . mem_k[key]   (2 lanes per key, 32 dims each)
    const _Float16* q = Qh + (size_t)(bi * NN + i) * INNER + h * DDIM + half * 32;
    const float* kp = mem_kv + base + (size_t)key * (2 * DDIM) + half * 32;
    float part = 0.0f;
    #pragma unroll
    for (int j4 = 0; j4 < 8; ++j4) {
        float4 kv4 = *reinterpret_cast<const float4*>(kp + j4 * 4);
        part += (float)q[j4 * 4 + 0] * kv4.x + (float)q[j4 * 4 + 1] * kv4.y
              + (float)q[j4 * 4 + 2] * kv4.z + (float)q[j4 * 4 + 3] * kv4.w;
    }
    part += __shfl_xor(part, 1);
    float sim = part * scale;
    if (!mem_mask[gw * KMEM + key]) sim = -FLT_MAX;

    float mx = sim;
    #pragma unroll
    for (int msk = 2; msk < 32; msk <<= 1) mx = fmaxf(mx, __shfl_xor(mx, msk));
    float p = (mx == -FLT_MAX) ? 0.0f : expf(sim - mx);
    float ls = p;
    #pragma unroll
    for (int msk = 2; msk < 32; msk <<= 1) ls += __shfl_xor(ls, msk);  // 16 distinct keys

    // A_mem[d] = sum_key p[key] * mem_v[key][d]; lane owns 2 features
    const int d0 = lane * 2;
    float ax = 0.0f, ay = 0.0f;
    #pragma unroll
    for (int kk = 0; kk < KMEM; ++kk) {
        float pk = __shfl(p, kk * 2);
        float2 vv = *reinterpret_cast<const float2*>(mem_kv + base + (size_t)kk * (2 * DDIM) + DDIM + d0);
        ax += pk * vv.x; ay += pk * vv.y;
    }
    float* arow = Acc + gw * DDIM;
    arow[d0] = ax; arow[d0 + 1] = ay;
    if (lane == 0) { Mst[gw] = mx; Lst[gw] = ls; }
}

// ---------------------------------------------------------------------------
// Merge local & memory branches via log-sum-exp; write f16 context (b,n,H*D).
// ---------------------------------------------------------------------------
__global__ __launch_bounds__(256) void merge_ctx(
    const float* __restrict__ Al, const float* __restrict__ Ml, const float* __restrict__ Ll,
    const float* __restrict__ Am, const float* __restrict__ Mm, const float* __restrict__ Lm,
    _Float16* __restrict__ Ctx)
{
    const int t = threadIdx.x, lane = t & 31, w = t >> 5;
    const size_t gw = (size_t)blockIdx.x * 8 + w;
    const int i  = (int)(gw % NN);
    const int bh = (int)(gw / NN);
    const int h  = bh % HH, bi = bh / HH;
    float ml = Ml[gw], ll = Ll[gw], mm = Mm[gw], lm2 = Lm[gw];
    float M  = fmaxf(ml, mm);
    float el = expf(ml - M), em = expf(mm - M);
    float invd = 1.0f / (ll * el + lm2 * em);
    const int d0 = lane * 2;
    const float* al = Al + gw * DDIM;
    const float* am = Am + gw * DDIM;
    float c0 = (al[d0] * el + am[d0] * em) * invd;
    float c1 = (al[d0 + 1] * el + am[d0 + 1] * em) * invd;
    size_t obase = (size_t)(bi * NN + i) * INNER + h * DDIM + d0;
    Ctx[obase]     = (_Float16)c0;
    Ctx[obase + 1] = (_Float16)c1;
}

// ---------------------------------------------------------------------------
extern "C" void kernel_launch(void* const* d_in, const int* in_sizes, int n_in,
                              void* d_out, int out_size, void* d_ws, size_t ws_size,
                              hipStream_t stream) {
    (void)in_sizes; (void)n_in; (void)out_size; (void)ws_size;
    const float* x           = (const float*)d_in[0];
    const float* mem_kv      = (const float*)d_in[1];
    const unsigned char* msk = (const unsigned char*)d_in[2];  // bool array (1 byte)
    const float* Wq          = (const float*)d_in[3];
    const float* Wkv         = (const float*)d_in[4];
    const float* Wo          = (const float*)d_in[5];
    const float* scale_param = (const float*)d_in[6];
    const float* gate        = (const float*)d_in[7];

    char* ws = (char*)d_ws;                          // ~27 MB total
    _Float16*  q_h = (_Float16*)(ws);                              // 4 MB
    _Float16*  k_h = (_Float16*)(ws + (4u  << 20));                // 0.5 MB
    _Float16*  v_h = (_Float16*)(ws + (4u  << 20) + (512u << 10)); // 0.5 MB
    float*     Al  = (float*)(ws + (5u  << 20));                   // 8 MB
    float*     Ml  = (float*)(ws + (13u << 20));                   // 128 KB
    float*     Ll  = (float*)(ws + (13u << 20) + (128u << 10));    // 128 KB
    float*     Am  = (float*)(ws + (14u << 20));                   // 8 MB
    float*     Mm  = (float*)(ws + (22u << 20));                   // 128 KB
    float*     Lm  = (float*)(ws + (22u << 20) + (128u << 10));    // 128 KB
    _Float16*  ctx = (_Float16*)(ws + (23u << 20));                // 4 MB

    const int rows = BB * NN;                        // 4096
    gemm_qnorm <<<dim3(INNER / 128, rows / 64), 256, 0, stream>>>(x, Wq,  q_h, rows, INNER, DIM);
    gemm_kvnorm<<<dim3(1,           rows / 64), 256, 0, stream>>>(x, Wkv, k_h, v_h, rows, 128, DIM);
    flash_local<<<dim3(NN / 128, BB * HH), 256, 0, stream>>>(q_h, k_h, v_h, scale_param, Al, Ml, Ll);
    mem_attn   <<<(BB * HH * NN) / 8, 256, 0, stream>>>(q_h, mem_kv, msk, scale_param, Am, Mm, Lm);
    merge_ctx  <<<(BB * HH * NN) / 8, 256, 0, stream>>>(Al, Ml, Ll, Am, Mm, Lm, ctx);
    gemm_out   <<<dim3(DIM / 128, rows / 64), 256, 0, stream>>>(ctx, Wo, x, gate, (float*)d_out, rows, DIM, INNER);
}
